// WindowAttention_46488726012581
// MI455X (gfx1250) — matrix operand order repaired and checked by reading, not compiled, not run on hardware
//
#include <hip/hip_runtime.h>
#include <hip/hip_bf16.h>

// ---------------------------------------------------------------------------
// Swin window attention, fused for MI455X (gfx1250, wave32, WMMA bf16).
//   B=4096 windows, N=49 tokens, C=384, H=12 heads, HD=32, NW=64 mask windows.
// Pipeline:
//   k0: convert w_qkv / w_proj f32 -> bf16 (workspace)
//   k1: per-window fused  qkv GEMM -> attention (softmax in-regs) -> bf16 out
//   k2: output projection GEMM (bf16 WMMA, f32 out + bias)
// GEMM loops ordered so every memory operand address varies with the inner
// loops (no loop-invariant global loads for LLVM to hoist/spill), and
// consecutive WMMAs hit independent accumulators (no D->C RAW stalls).
// ---------------------------------------------------------------------------

typedef __bf16 bf16_t;
typedef __attribute__((ext_vector_type(16))) __bf16 v16bf;
typedef __attribute__((ext_vector_type(8)))  float  v8f;

union Frag16 {
  v16bf v;
  uint4 u[2];
};

union Pack4 {
  bf16_t h[4];
  uint2  u;
};

__device__ __forceinline__ v8f wmma_bf16(v16bf a, v16bf b, v8f c) {
  // D = A(16x32 bf16) * B(32x16 bf16) + C(16x16 f32)
  return __builtin_amdgcn_wmma_f32_16x16x32_bf16(false, a, false, b,
                                                 (short)0, c, false, false);
}

#define B_TOT 4096
#define N_TOK 49
#define C_DIM 384
#define H_CNT 12
#define HD_   32
#define QKV_N 1152

// LDS strides (halves), padded to avoid 64-bank conflicts
#define XS_STRIDE 392   // x tile rows (64 x 392)
#define QK_STRIDE 40    // q/k per-head rows (64 x 40)
#define VT_STRIDE 72    // v transposed rows (32 x 72) per head
#define PS_STRIDE 72    // softmax P scratch rows (16 x 72) per wave

#define XS_HALVES (64 * XS_STRIDE)                 // 25088
#define QK_HALVES (2 * H_CNT * 64 * QK_STRIDE)     // 61440
#define VT_HALVES (H_CNT * HD_ * VT_STRIDE)        // 27648
#define PS_HALVES (8 * 16 * PS_STRIDE)             // 9216
#define SMEM_BYTES ((XS_HALVES + QK_HALVES + VT_HALVES + PS_HALVES) * 2)

// ---------------------------------------------------------------------------
__global__ void cvt_weights(const float* __restrict__ wqkv,
                            const float* __restrict__ wproj,
                            bf16_t* __restrict__ wqkv_bf,
                            bf16_t* __restrict__ wproj_bf) {
  const int nq = QKV_N * C_DIM;   // 442368
  const int np = C_DIM * C_DIM;   // 147456
  int i = blockIdx.x * blockDim.x + threadIdx.x;
  if (i < nq)            wqkv_bf[i]       = (bf16_t)wqkv[i];
  else if (i < nq + np)  wproj_bf[i - nq] = (bf16_t)wproj[i - nq];
}

// ---------------------------------------------------------------------------
// One workgroup per window: qkv projection + masked softmax attention.
__global__ __launch_bounds__(256, 4)   // 2nd arg: min waves/EU -> ~256 VGPR cap
void fused_qkv_attn(const float* __restrict__ x,
                    const float* __restrict__ mask,
                    const bf16_t* __restrict__ wqkv_bf,
                    const float* __restrict__ bqkv,
                    bf16_t* __restrict__ attn_out) {
  extern __shared__ char smem_raw[];
  bf16_t* xs = (bf16_t*)smem_raw;          // [64][392] bf16 x tile
  bf16_t* qk = xs + XS_HALVES;             // [2][12][64][40] q,k
  bf16_t* vT = qk + QK_HALVES;             // [12][32][72]   v transposed
  bf16_t* ps = vT + VT_HALVES;             // [8][16][72]    P scratch per wave

  const int b    = blockIdx.x;
  const int tid  = threadIdx.x;
  const int lane = tid & 31;
  const int wv   = tid >> 5;       // 8 waves
  const int lm   = lane & 15;      // lane within half-wave
  const int lh   = lane >> 4;      // half-wave (ISA K/M split)

  // ---- Phase 0: stage x[b] -> LDS bf16 (4-wide packed), zero rows 49..63 ----
  const float* xb = x + (long)b * N_TOK * C_DIM;
  for (int i = tid; i < 64 * (C_DIM / 4); i += 256) {
    int r = i / (C_DIM / 4), c4 = (i - r * (C_DIM / 4)) * 4;
    Pack4 p;
    if (r < N_TOK) {
      float4 f = *(const float4*)(xb + r * C_DIM + c4);
      p.h[0] = (bf16_t)f.x; p.h[1] = (bf16_t)f.y;
      p.h[2] = (bf16_t)f.z; p.h[3] = (bf16_t)f.w;
    } else {
      p.u = make_uint2(0u, 0u);
    }
    *(uint2*)(xs + r * XS_STRIDE + c4) = p.u;
  }
  __syncthreads();

  // ---- Phase 1: qkv = xs @ wqkv^T + b  (M=64, K=384, N=1152) ----
  // wave wv owns output columns [wv*144, wv*144+144) = 9 tiles of 16.
  // nt outer, k middle, mt inner: B frag loaded once and consumed by 4
  // independent accumulator chains; no loop-invariant global loads remain.
  const int colbase = wv * 144;
  for (int nt = 0; nt < 9; ++nt) {
    asm volatile("" ::: "memory");         // no cross-iteration load caching
    v8f acc[4] = {};
    const bf16_t* brow0 =
        wqkv_bf + (long)(colbase + nt * 16 + lm) * C_DIM + (lh ? 16 : 0);
    for (int k = 0; k < 12; ++k) {
      Frag16 bfr;                          // B layout: 16 contiguous K halves
      bfr.u[0] = *(const uint4*)(brow0 + k * 32);
      bfr.u[1] = *(const uint4*)(brow0 + k * 32 + 8);
#pragma unroll
      for (int mt = 0; mt < 4; ++mt) {
        const bf16_t* arow =
            xs + (mt * 16 + lm) * XS_STRIDE + k * 32 + (lh ? 16 : 0);
        Frag16 a;                          // A layout: 16 contiguous K halves
        a.u[0] = *(const uint4*)(arow);
        a.u[1] = *(const uint4*)(arow + 8);
        acc[mt] = wmma_bf16(a.v, bfr.v, acc[mt]);
      }
    }
    const int col = colbase + nt * 16 + lm;      // output column == wqkv row
    const float bias = bqkv[col];
    const int s   = col / C_DIM;           // 0=q 1=k 2=v (uniform per tile)
    const int rem = col - s * C_DIM;
    const int h   = rem >> 5;
    const int d   = rem & 31;
#pragma unroll
    for (int mt = 0; mt < 4; ++mt) {
#pragma unroll
      for (int j = 0; j < 8; ++j) {        // C layout: M = j + lh*8
        int m = mt * 16 + j + lh * 8;
        bf16_t val = (bf16_t)(acc[mt][j] + bias);
        if (s == 2) vT[(h * HD_ + d) * VT_STRIDE + m] = val;       // [h][d][n]
        else        qk[((s * H_CNT + h) * 64 + m) * QK_STRIDE + d] = val;
      }
    }
  }
  __syncthreads();

  // ---- Phase 2: per-head attention ----
  const float scale = 0.17677669529663687f;          // 1/sqrt(32)
  const float* maskw = mask + (long)(b & 63) * N_TOK * N_TOK;
  bf16_t* psw = ps + wv * 16 * PS_STRIDE;

  for (int h = wv; h < H_CNT; h += 8) {
    const bf16_t* qh = qk + (0 * H_CNT + h) * 64 * QK_STRIDE;
    const bf16_t* kh = qk + (1 * H_CNT + h) * 64 * QK_STRIDE;
    const bf16_t* vh = vT + h * HD_ * VT_STRIDE;

    for (int mt = 0; mt < 4; ++mt) {
      // S(16x64) = q_tile(16x32) @ k^T(32x64), one K=32 step
      Frag16 aq;
      const bf16_t* qrow = qh + (mt * 16 + lm) * QK_STRIDE + (lh ? 16 : 0);
      aq.u[0] = *(const uint4*)(qrow);
      aq.u[1] = *(const uint4*)(qrow + 8);
      v8f sacc[4];
#pragma unroll
      for (int nt = 0; nt < 4; ++nt) {
        Frag16 bk;   // B col n = k row n; halves = d contiguous
        const bf16_t* krow = kh + (nt * 16 + lm) * QK_STRIDE + (lh ? 16 : 0);
        bk.u[0] = *(const uint4*)(krow);
        bk.u[1] = *(const uint4*)(krow + 8);
        v8f z = {};
        sacc[nt] = wmma_bf16(aq.v, bk.v, z);
      }
      // masked softmax over the 64 (padded) columns; row split across 16 lanes
#pragma unroll
      for (int j = 0; j < 8; ++j) {
        int r  = mt * 16 + j + lh * 8;
        int rm = (r < N_TOK) ? r : (N_TOK - 1);       // clamp: no OOB mask read
        float vv[4];
#pragma unroll
        for (int nt = 0; nt < 4; ++nt) {
          int c = nt * 16 + lm;
          float mv = (c < N_TOK) ? maskw[rm * N_TOK + c] : -1e30f;
          vv[nt] = sacc[nt][j] * scale + mv;
        }
        float mx = fmaxf(fmaxf(vv[0], vv[1]), fmaxf(vv[2], vv[3]));
#pragma unroll
        for (int off = 1; off < 16; off <<= 1)        // stays in half-wave
          mx = fmaxf(mx, __shfl_xor(mx, off, 32));
        float e[4], sm = 0.0f;
#pragma unroll
        for (int nt = 0; nt < 4; ++nt) { e[nt] = __expf(vv[nt] - mx); sm += e[nt]; }
#pragma unroll
        for (int off = 1; off < 16; off <<= 1)
          sm += __shfl_xor(sm, off, 32);
        float inv = 1.0f / sm;
#pragma unroll
        for (int nt = 0; nt < 4; ++nt)
          psw[(j + lh * 8) * PS_STRIDE + nt * 16 + lm] = (bf16_t)(e[nt] * inv);
      }
      asm volatile("s_wait_dscnt 0x0" ::: "memory");  // same-wave P store->load

      // O(16x32) = P(16x64) @ V(64x32), two K=32 steps; V is pre-transposed
#pragma unroll
      for (int dt = 0; dt < 2; ++dt) {
        v8f oacc = {};
#pragma unroll
        for (int kk = 0; kk < 2; ++kk) {
          Frag16 ap, bv;
          const bf16_t* prow = psw + lm * PS_STRIDE + kk * 32 + (lh ? 16 : 0);
          ap.u[0] = *(const uint4*)(prow);
          ap.u[1] = *(const uint4*)(prow + 8);
          const bf16_t* vrow = vh + (dt * 16 + lm) * VT_STRIDE + kk * 32 + (lh ? 16 : 0);
          bv.u[0] = *(const uint4*)(vrow);
          bv.u[1] = *(const uint4*)(vrow + 8);
          oacc = wmma_bf16(ap.v, bv.v, oacc);
        }
#pragma unroll
        for (int j = 0; j < 8; ++j) {
          int m = mt * 16 + j + lh * 8;
          if (m < N_TOK)
            attn_out[((long)b * N_TOK + m) * C_DIM + h * HD_ + dt * 16 + lm] =
                (bf16_t)oacc[j];
        }
      }
    }
  }
}

// ---------------------------------------------------------------------------
// Output projection: out = attn(200704x384) @ w_proj^T(384x384) + b_proj (f32)
// Wave strip: 16 rows x 192 cols (12 resident accumulators), k-outer.
// Workgroup: 4 row strips x 2 column halves (64 rows x 384 cols).
__global__ __launch_bounds__(256, 4)
void proj_kernel(const bf16_t* __restrict__ attn,
                 const bf16_t* __restrict__ wproj_bf,
                 const float* __restrict__ bproj,
                 float* __restrict__ out) {
  const int tid  = threadIdx.x;
  const int lane = tid & 31;
  const int wv   = tid >> 5;
  const int lm   = lane & 15;
  const int lh   = lane >> 4;
  const long rbase = (long)blockIdx.x * 64 + (wv >> 1) * 16;
  const int  ch    = wv & 1;                     // column half: cols [ch*192..)

  const bf16_t* arow = attn + (rbase + lm) * C_DIM + (lh ? 16 : 0);
  const bf16_t* bbase = wproj_bf + (long)(ch * 192 + lm) * C_DIM + (lh ? 16 : 0);

  v8f acc[12] = {};
  for (int k = 0; k < 12; ++k) {
    Frag16 a;
    a.u[0] = *(const uint4*)(arow + k * 32);
    a.u[1] = *(const uint4*)(arow + k * 32 + 8);
#pragma unroll
    for (int t = 0; t < 12; ++t) {
      const bf16_t* brow = bbase + (long)t * 16 * C_DIM + k * 32;
      Frag16 bfr;
      bfr.u[0] = *(const uint4*)(brow);
      bfr.u[1] = *(const uint4*)(brow + 8);
      acc[t] = wmma_bf16(a.v, bfr.v, acc[t]);
    }
  }
#pragma unroll
  for (int t = 0; t < 12; ++t) {
    const int col = ch * 192 + t * 16 + lm;
    const float bias = bproj[col];
#pragma unroll
    for (int j = 0; j < 8; ++j)
      out[(rbase + j + lh * 8) * C_DIM + col] = acc[t][j] + bias;
  }
}

// ---------------------------------------------------------------------------
extern "C" void kernel_launch(void* const* d_in, const int* in_sizes, int n_in,
                              void* d_out, int out_size, void* d_ws, size_t ws_size,
                              hipStream_t stream) {
  (void)in_sizes; (void)n_in; (void)out_size; (void)ws_size;
  const float* x      = (const float*)d_in[0];
  const float* mask   = (const float*)d_in[1];
  const float* w_qkv  = (const float*)d_in[2];
  const float* b_qkv  = (const float*)d_in[3];
  const float* w_proj = (const float*)d_in[4];
  const float* b_proj = (const float*)d_in[5];
  float* out = (float*)d_out;

  // workspace: [bf16 w_qkv | bf16 w_proj | pad | bf16 attn_out (B,N,C)]
  const size_t nq = (size_t)QKV_N * C_DIM;            // 442368
  const size_t np = (size_t)C_DIM * C_DIM;            // 147456
  bf16_t* wqkv_bf  = (bf16_t*)d_ws;
  bf16_t* wproj_bf = wqkv_bf + nq;
  bf16_t* attn_bf  = (bf16_t*)((char*)d_ws + 1179648); // 256B-aligned, ~147MB

  const int cvt_blocks = (int)((nq + np + 255) / 256);
  cvt_weights<<<cvt_blocks, 256, 0, stream>>>(w_qkv, w_proj, wqkv_bf, wproj_bf);

  fused_qkv_attn<<<B_TOT, 256, SMEM_BYTES, stream>>>(x, mask, wqkv_bf, b_qkv,
                                                     attn_bf);

  proj_kernel<<<(B_TOT * N_TOK) / 64, 256, 0, stream>>>(attn_bf, wproj_bf,
                                                        b_proj, out);
}
